// Paramerized_StepwiseGRU_22892175687867
// MI455X (gfx1250) — matrix operand order repaired and checked by reading, not compiled, not run on hardware
//
#include <hip/hip_runtime.h>
#include <hip/hip_bf16.h>

typedef __attribute__((ext_vector_type(16))) _Float16 v16h;
typedef __attribute__((ext_vector_type(8)))  float    v8f;
typedef __attribute__((ext_vector_type(4)))  float    v4f;

#define DD 128
#define BROWS 524288
#define NTILES_ROW (BROWS / 128)       // 4096 row-tiles of 128 rows
#define NUM_WG 512
#define WG_THREADS 256
#define WFRAG_BYTES (6 * 32 * 1024)    // 6 gates * 32 frags * 1KB = 196608
#define BIAS_BYTES (4 * DD * 4)        // 2KB fused biases
#define BIAS_OFF WFRAG_BYTES
#define SMEM_BYTES (WFRAG_BYTES + BIAS_BYTES)

static __device__ __forceinline__ v8f wmma_f16(v16h a, v16h b, v8f c) {
    // 8-arg form: (neg_a, A, neg_b, B, c_mod, C, reuse_a, reuse_b)
    return __builtin_amdgcn_wmma_f32_16x16x32_f16(false, a, false, b, (short)0, c, false, false);
}

static __device__ __forceinline__ float fast_sigmoid(float v) {
    return 1.0f / (1.0f + __expf(-v));
}

static __device__ __forceinline__ float fast_tanh(float v) {
    v = fminf(fmaxf(v, -15.0f), 15.0f);
#if __has_builtin(__builtin_amdgcn_tanhf)
    return __builtin_amdgcn_tanhf(v);
#elif __has_builtin(__builtin_amdgcn_tanh_f32)
    return __builtin_amdgcn_tanh_f32(v);
#else
    float e = __expf(2.0f * v);
    return (e - 1.0f) / (e + 1.0f);
#endif
}

// ---------------------------------------------------------------------------
// Kernel 1: repack 6 fp32 weight matrices (128x128, W[k][n]) into f16 WMMA
// B-fragments in d_ws, plus fused biases.
// B-fragment layout (16-bit B 32x16, wave32):
//   lanes 0-15  : N = lane,      elements j=0..15 hold K = kt*32 + j
//   lanes 16-31 : N = lane-16,   elements j=0..15 hold K = kt*32 + 16 + j
// Stored per fragment as frag[lane][16] halfs -> lane reads 32 contiguous B.
// ---------------------------------------------------------------------------
__global__ void gru_repack(const float* __restrict__ Wzi, const float* __restrict__ Wzm,
                           const float* __restrict__ Wri, const float* __restrict__ Wrm,
                           const float* __restrict__ Wci, const float* __restrict__ Wcm,
                           const float* __restrict__ bzi, const float* __restrict__ bzm,
                           const float* __restrict__ bri, const float* __restrict__ brm,
                           const float* __restrict__ bci, const float* __restrict__ bcm,
                           unsigned char* __restrict__ ws)
{
    int tid = blockIdx.x * blockDim.x + threadIdx.x;   // 0 .. 98303
    int g   = tid >> 14;                               // gate 0..5
    int rem = tid & 16383;
    int f   = rem >> 9;                                // fragment 0..31
    int e   = rem & 511;
    int kt  = f >> 3, nt = f & 7;
    int l   = e >> 4, j  = e & 15;
    int k   = kt * 32 + ((l >> 4) << 4) + j;
    int n   = nt * 16 + (l & 15);
    const float* W = (g == 0) ? Wzi : (g == 1) ? Wzm : (g == 2) ? Wri
                   : (g == 3) ? Wrm : (g == 4) ? Wci : Wcm;
    _Float16* dsth = (_Float16*)ws;
    dsth[tid] = (_Float16)W[k * DD + n];

    if (blockIdx.x == 0 && threadIdx.x < DD) {
        float* bias = (float*)(ws + BIAS_OFF);
        int c = threadIdx.x;
        bias[c]          = bzi[c] + bzm[c];
        bias[DD + c]     = bri[c] + brm[c];
        bias[2 * DD + c] = bci[c];
        bias[3 * DD + c] = bcm[c];
    }
}

// ---------------------------------------------------------------------------
// Kernel 2: main GRU step.
// Workgroup = 8 waves; wave w owns rows [tile*128 + w*16, +16).
// A fragments (x/h, hi/lo, 4 ktiles = 128 VGPRs) built once per tile.
// N-tile loop keeps 4 accumulators live; B fragments double-buffered over kt;
// WMMAs round-robin the 4 accumulator chains; sched_group_barrier requests
// a 6xDS-read / 6xWMMA interleave so LDS runs ahead of the matrix pipe.
// ---------------------------------------------------------------------------
__global__ __launch_bounds__(WG_THREADS, 1)
void gru_main(const float* __restrict__ x, const float* __restrict__ h,
              const unsigned char* __restrict__ ws, float* __restrict__ out)
{
    extern __shared__ unsigned char smem[];

    // Stage 192KB weight fragments + 2KB biases: global -> LDS (b128 copies).
    {
        const uint4* src = (const uint4*)ws;
        uint4*       dst = (uint4*)smem;
        for (int i = threadIdx.x; i < SMEM_BYTES / 16; i += WG_THREADS)
            dst[i] = src[i];
    }
    __syncthreads();

    const int lane     = threadIdx.x & 31;
    const int wave     = threadIdx.x >> 5;
    const int laneM    = lane & 15;
    const int laneHalf = lane >> 4;        // 0 or 1
    const int laneKA   = laneHalf * 16;    // A-fragment K offset

    const _Float16* wf    = (const _Float16*)smem;
    const float*    biasL = (const float*)(smem + BIAS_OFF);

    for (int tile = blockIdx.x; tile < NTILES_ROW; tile += gridDim.x) {
        const int r0 = tile * 128 + wave * 16;
        const float* xr = x + (size_t)(r0 + laneM) * DD;
        const float* hr = h + (size_t)(r0 + laneM) * DD;

        // Prefetch next tile's rows (global_prefetch_b8)
        int ptile = tile + gridDim.x;
        if (ptile < NTILES_ROW) {
            const float* px = x + (size_t)(ptile * 128 + wave * 16 + laneM) * DD;
            const float* ph = h + (size_t)(ptile * 128 + wave * 16 + laneM) * DD;
            __builtin_prefetch((const void*)px, 0, 1);
            __builtin_prefetch((const void*)ph, 0, 1);
        }

        // Build ALL A fragments up front: x/h, hi/lo, 4 ktiles (128 VGPRs).
        v16h AXH[4], AXL[4], AHH[4], AHL[4];
#pragma unroll
        for (int kt = 0; kt < 4; ++kt) {
            const int c0 = kt * 32 + laneKA;
            v4f X0 = *(const v4f*)(xr + c0);
            v4f X1 = *(const v4f*)(xr + c0 + 4);
            v4f X2 = *(const v4f*)(xr + c0 + 8);
            v4f X3 = *(const v4f*)(xr + c0 + 12);
            v4f H0 = *(const v4f*)(hr + c0);
            v4f H1 = *(const v4f*)(hr + c0 + 4);
            v4f H2 = *(const v4f*)(hr + c0 + 8);
            v4f H3 = *(const v4f*)(hr + c0 + 12);
#pragma unroll
            for (int j = 0; j < 4; ++j) {
                float v;
                _Float16 hi;
                v = X0[j]; hi = (_Float16)v; AXH[kt][j]      = hi; AXL[kt][j]      = (_Float16)(v - (float)hi);
                v = X1[j]; hi = (_Float16)v; AXH[kt][j + 4]  = hi; AXL[kt][j + 4]  = (_Float16)(v - (float)hi);
                v = X2[j]; hi = (_Float16)v; AXH[kt][j + 8]  = hi; AXL[kt][j + 8]  = (_Float16)(v - (float)hi);
                v = X3[j]; hi = (_Float16)v; AXH[kt][j + 12] = hi; AXL[kt][j + 12] = (_Float16)(v - (float)hi);
                v = H0[j]; hi = (_Float16)v; AHH[kt][j]      = hi; AHL[kt][j]      = (_Float16)(v - (float)hi);
                v = H1[j]; hi = (_Float16)v; AHH[kt][j + 4]  = hi; AHL[kt][j + 4]  = (_Float16)(v - (float)hi);
                v = H2[j]; hi = (_Float16)v; AHH[kt][j + 8]  = hi; AHL[kt][j + 8]  = (_Float16)(v - (float)hi);
                v = H3[j]; hi = (_Float16)v; AHH[kt][j + 12] = hi; AHL[kt][j + 12] = (_Float16)(v - (float)hi);
            }
        }

        // N-tile loop: 4 accumulators; B fragments double-buffered over kt.
#pragma unroll 1
        for (int nt = 0; nt < 8; ++nt) {
            v8f aZ = {}, aR = {}, aC = {}, aM = {};

            v16h B[2][6];
            {
                const int fo = nt * 512 + lane * 16;   // kt = 0
#pragma unroll
                for (int g = 0; g < 6; ++g)
                    B[0][g] = *(const v16h*)(wf + g * 16384 + fo);
            }

#pragma unroll
            for (int kt = 0; kt < 4; ++kt) {
                const int cb = kt & 1;
                const int nb = cb ^ 1;
                if (kt < 3) {
                    const int fo = ((kt + 1) * 8 + nt) * 512 + lane * 16;
#pragma unroll
                    for (int g = 0; g < 6; ++g)
                        B[nb][g] = *(const v16h*)(wf + g * 16384 + fo);
                }

                // Round-robin the 4 independent accumulator chains so
                // adjacent WMMAs are independent.
                aZ = wmma_f16(AXL[kt], B[cb][0], aZ);
                aR = wmma_f16(AXL[kt], B[cb][2], aR);
                aC = wmma_f16(AXL[kt], B[cb][4], aC);
                aM = wmma_f16(AHL[kt], B[cb][5], aM);

                aZ = wmma_f16(AXH[kt], B[cb][0], aZ);
                aR = wmma_f16(AXH[kt], B[cb][2], aR);
                aC = wmma_f16(AXH[kt], B[cb][4], aC);
                aM = wmma_f16(AHH[kt], B[cb][5], aM);

                aZ = wmma_f16(AHL[kt], B[cb][1], aZ);
                aR = wmma_f16(AHL[kt], B[cb][3], aR);
                aZ = wmma_f16(AHH[kt], B[cb][1], aZ);
                aR = wmma_f16(AHH[kt], B[cb][3], aR);
            }

#if __has_builtin(__builtin_amdgcn_sched_group_barrier)
            // Request: alternate 6 LDS reads with 6 WMMAs, 8 times
            // (48 ds_load + 48 wmma in this region).
#pragma unroll
            for (int sg = 0; sg < 8; ++sg) {
                __builtin_amdgcn_sched_group_barrier(0x100, 6, 0); // DS read
                __builtin_amdgcn_sched_group_barrier(0x008, 6, 0); // WMMA/MFMA
            }
#endif

            // Epilogue for this nt. C/D layout: VGPR j, lanes 0-15 -> row j,
            // lanes 16-31 -> row j+8; col = nt*16 + (lane&15).
            const int col = nt * 16 + laneM;
            const float bzv = biasL[col];
            const float brv = biasL[DD + col];
            const float bcv = biasL[2 * DD + col];
            const float bmv = biasL[3 * DD + col];
#pragma unroll
            for (int j = 0; j < 8; ++j) {
                const int row = j + laneHalf * 8;
                const size_t idx = (size_t)(r0 + row) * DD + col;
                float hv = h[idx];
                float zv = fast_sigmoid(aZ[j] + bzv);
                float rv = fast_sigmoid(aR[j] + brv);
                float cv = fast_tanh(aC[j] + bcv + rv * (aM[j] + bmv));
                out[idx] = zv * hv + (1.0f - zv) * cv;
            }
        }
    }
}

extern "C" void kernel_launch(void* const* d_in, const int* in_sizes, int n_in,
                              void* d_out, int out_size, void* d_ws, size_t ws_size,
                              hipStream_t stream) {
    (void)in_sizes; (void)n_in; (void)out_size; (void)ws_size;
    const float* x   = (const float*)d_in[0];
    const float* h   = (const float*)d_in[1];
    const float* Wzi = (const float*)d_in[2];
    const float* bzi = (const float*)d_in[3];
    const float* Wzm = (const float*)d_in[4];
    const float* bzm = (const float*)d_in[5];
    const float* Wri = (const float*)d_in[6];
    const float* bri = (const float*)d_in[7];
    const float* Wrm = (const float*)d_in[8];
    const float* brm = (const float*)d_in[9];
    const float* Wci = (const float*)d_in[10];
    const float* bci = (const float*)d_in[11];
    const float* Wcm = (const float*)d_in[12];
    const float* bcm = (const float*)d_in[13];
    float* out = (float*)d_out;
    unsigned char* ws = (unsigned char*)d_ws;

    // Allow ~194KB dynamic LDS (CDNA5 WGP supports up to 320KB per workgroup).
    (void)hipFuncSetAttribute((const void*)gru_main,
                              hipFuncAttributeMaxDynamicSharedMemorySize,
                              SMEM_BYTES);

    gru_repack<<<dim3((6 * 16384) / WG_THREADS), dim3(WG_THREADS), 0, stream>>>(
        Wzi, Wzm, Wri, Wrm, Wci, Wcm, bzi, bzm, bri, brm, bci, bcm, ws);

    gru_main<<<dim3(NUM_WG), dim3(WG_THREADS), SMEM_BYTES, stream>>>(x, h, ws, out);
}
